// DeepseekV32Indexer_18545668784294
// MI455X (gfx1250) — matrix operand order repaired and checked by reading, not compiled, not run on hardware
//
#include <hip/hip_runtime.h>
#include <hip/hip_bf16.h>

typedef __attribute__((ext_vector_type(16))) __bf16 v16bf;
typedef __attribute__((ext_vector_type(8)))  float  v8f;

#define S_LEN 2048
#define NHEAD 32
#define DHEAD 128
#define HIDDEN 4096
#define QLRANK 1536

union bf16x4 { __bf16 h[4]; uint2 u; };

__device__ __forceinline__ void store_bf16x4(__bf16* dst, float4 v4) {
  bf16x4 pk;
  pk.h[0] = (__bf16)v4.x; pk.h[1] = (__bf16)v4.y;
  pk.h[2] = (__bf16)v4.z; pk.h[3] = (__bf16)v4.w;
  *(uint2*)dst = pk.u;
}

// ---------------------------------------------------------------------------
// Generic GEMM: C[M x N] = scale * (A[M x K] @ B[N x K]^T), f32 in, f32 out.
// bf16 WMMA (v_wmma_f32_16x16x32_bf16), 8 waves/block, 16x128 tile, BK=64.
// Double-buffered LDS pipeline: next tile's global_load_b128s are issued
// between the barrier and the WMMAs so s_wait_loadcnt lands after the matrix
// ops (HBM latency hidden behind the WMMA pipe); one barrier per K-step.
// Block-uniform fullN fast path avoids exec-mask guards in the big GEMMs.
// Fragment layouts per CDNA5 ISA 7.12.2 (wave32):
//   A 16x32 bf16 : lane%16 = row M; lanes<16 hold K {0..7,16..23},
//                  lanes>=16 hold K {8..15,24..31} (2 per VGPR)
//   B 32x16 bf16 : lane%16 = col N; lanes<16 hold K {0..7,16..23}, lanes>=16 K {8..15,24..31}
//   C/D 16x16 f32: elem e -> row M = e + 8*(lane>=16), col N = lane%16
// ---------------------------------------------------------------------------
__global__ __launch_bounds__(256) void gemm_bf16_nt(
    const float* __restrict__ A, const float* __restrict__ B,
    float* __restrict__ C, int M, int N, int K, float scale)
{
  __shared__ __bf16 As[2][16][64];    //  4 KB
  __shared__ __bf16 Bs[2][128][64];   // 32 KB

  const int tid  = threadIdx.x;
  const int wave = tid >> 5;
  const int lane = tid & 31;
  const int lh   = lane >> 4;   // lane-half
  const int l16  = lane & 15;
  const int m0   = blockIdx.x * 16;
  const int n0   = blockIdx.y * 128;
  const int nCol = n0 + wave * 16 + l16;
  const bool fullN = (n0 + 128 <= N);

  const int ar = tid >> 4;            // A-tile row, col for this thread
  const int ac = (tid & 15) << 2;

  float4 aReg;
  float4 bReg[8];

  auto loadTiles = [&](int kk) {
    aReg = *(const float4*)&A[(size_t)(m0 + ar) * K + kk + ac];
    if (fullN) {
#pragma unroll
      for (int u = 0; u < 8; ++u) {
        int i = tid + u * 256;
        int r = i >> 4, c = (i & 15) << 2;
        bReg[u] = *(const float4*)&B[(size_t)(n0 + r) * K + kk + c];
      }
    } else {
#pragma unroll
      for (int u = 0; u < 8; ++u) {
        int i = tid + u * 256;
        int r = i >> 4, c = (i & 15) << 2;
        bReg[u] = make_float4(0.f, 0.f, 0.f, 0.f);
        if (n0 + r < N)
          bReg[u] = *(const float4*)&B[(size_t)(n0 + r) * K + kk + c];
      }
    }
  };

  loadTiles(0);
  int buf = 0;
  v8f acc = {};

  for (int kk = 0; kk < K; kk += 64) {
    // Commit staged registers to LDS (packed bf16, 64-bit stores)
    store_bf16x4(&As[buf][ar][ac], aReg);
#pragma unroll
    for (int u = 0; u < 8; ++u) {
      int i = tid + u * 256;
      int r = i >> 4, c = (i & 15) << 2;
      store_bf16x4(&Bs[buf][r][c], bReg[u]);
    }
    __syncthreads();

    if (kk + 64 < K) loadTiles(kk + 64);   // overlap with WMMAs below

#pragma unroll
    for (int sub = 0; sub < 64; sub += 32) {
      v16bf af, bfr;
#pragma unroll
      for (int e = 0; e < 16; ++e) {
        int ka = (e < 8 ? e : e + 8) + lh * 8;
        af[e]  = As[buf][l16][sub + ka];
        bfr[e] = Bs[buf][wave * 16 + l16][sub + ka];
      }
      acc = __builtin_amdgcn_wmma_f32_16x16x32_bf16(false, af, false, bfr,
                                                    (short)0, acc, false, false);
    }
    buf ^= 1;
  }

  if (nCol < N) {
#pragma unroll
    for (int e = 0; e < 8; ++e)
      C[(size_t)(m0 + e + lh * 8) * N + nCol] = acc[e] * scale;
  }
}

// ---------------------------------------------------------------------------
// k epilogue: layernorm(128) -> RoPE (last 64 dims, interleaved pairs) ->
// bf16 quantize -> FWHT(128) with bf16 rounding per stage -> * 128^-0.5
// One 128-thread block per sequence position.
// ---------------------------------------------------------------------------
__global__ __launch_bounds__(128) void k_epilogue(
    const float* __restrict__ kf, const float* __restrict__ freqs,
    const float* __restrict__ gamma, const float* __restrict__ beta,
    __bf16* __restrict__ kb)
{
  __shared__ float val[128];
  __shared__ float red[128];
  const int s = blockIdx.x, t = threadIdx.x;

  float v = kf[(size_t)s * 128 + t];
  red[t] = v; __syncthreads();
  for (int o = 64; o > 0; o >>= 1) { if (t < o) red[t] += red[t + o]; __syncthreads(); }
  float mu = red[0] * (1.0f / 128.0f);
  __syncthreads();
  float d = v - mu;
  red[t] = d * d; __syncthreads();
  for (int o = 64; o > 0; o >>= 1) { if (t < o) red[t] += red[t + o]; __syncthreads(); }
  float var = red[0] * (1.0f / 128.0f);
  float n = d * rsqrtf(var + 1e-5f) * gamma[t] + beta[t];

  val[t] = n; __syncthreads();
  float r = n;
  if (t >= 64) {              // RoPE on pe half: pairs (64+2p, 64+2p+1)
    int base = t & ~1;
    int p = (t - 64) >> 1;
    float f = freqs[s * 32 + p];
    float c = cosf(f), si = sinf(f);
    float x1 = val[base], x2 = val[base + 1];
    r = (t & 1) ? (x1 * si + x2 * c) : (x1 * c - x2 * si);
  }
  __syncthreads();
  val[t] = (float)(__bf16)r;  // quantize before FWHT, as reference does
  __syncthreads();

  for (int h = 1; h < 128; h <<= 1) {
    float a = val[t], b = val[t ^ h];
    __syncthreads();
    float nv = (t & h) ? (b - a) : (a + b);
    val[t] = (float)(__bf16)nv;   // bf16 arithmetic per stage
    __syncthreads();
  }
  kb[(size_t)s * 128 + t] = (__bf16)(val[t] * 0.08838834764831845f); // 128^-0.5
}

// ---------------------------------------------------------------------------
// q epilogue: RoPE + FWHT per (s, h) row of 128.  q_f32 is (S, H*D) row-major.
// ---------------------------------------------------------------------------
__global__ __launch_bounds__(128) void q_epilogue(
    const float* __restrict__ qf, const float* __restrict__ freqs,
    __bf16* __restrict__ qb)
{
  __shared__ float val[128];
  const int s = blockIdx.x, h = blockIdx.y, t = threadIdx.x;

  float v = qf[(size_t)s * (NHEAD * DHEAD) + h * DHEAD + t];
  val[t] = v; __syncthreads();
  float r = v;
  if (t >= 64) {
    int base = t & ~1;
    int p = (t - 64) >> 1;
    float f = freqs[s * 32 + p];
    float c = cosf(f), si = sinf(f);
    float x1 = val[base], x2 = val[base + 1];
    r = (t & 1) ? (x1 * si + x2 * c) : (x1 * c - x2 * si);
  }
  __syncthreads();
  val[t] = (float)(__bf16)r;
  __syncthreads();

  for (int hs = 1; hs < 128; hs <<= 1) {
    float a = val[t], b = val[t ^ hs];
    __syncthreads();
    float nv = (t & hs) ? (b - a) : (a + b);
    val[t] = (float)(__bf16)nv;
    __syncthreads();
  }
  qb[((size_t)s * NHEAD + h) * DHEAD + t] =
      (__bf16)(val[t] * 0.08838834764831845f);
}

// ---------------------------------------------------------------------------
// Score kernel: per block a 16(q) x 128(k) output tile.
// K tile (128 rows x 128 d, bf16, 32 KB) is head-invariant -> loaded once and
// reused across all 32 heads.  Per-head Q tile is double-buffered: next
// head's uint4 loads are issued between the barrier and this head's 4 chained
// WMMAs.  relu * weights[q,h] accumulated in f32 across heads.
// ---------------------------------------------------------------------------
__global__ __launch_bounds__(256) void scores_kernel(
    const __bf16* __restrict__ qb,   // (S, H, 128)
    const __bf16* __restrict__ kb,   // (S, 128)
    const float*  __restrict__ wts,  // (S, H)
    float* __restrict__ scores)      // (S, S)
{
  __shared__ __bf16 Ks[128][128];     // 32 KB
  __shared__ __bf16 Qs[2][16][128];   //  8 KB
  __shared__ float  Ws[16][32];       //  2 KB

  const int tid  = threadIdx.x;
  const int wave = tid >> 5;
  const int lane = tid & 31;
  const int lh   = lane >> 4;
  const int l16  = lane & 15;
  const int q0   = blockIdx.x * 16;
  const int k0   = blockIdx.y * 128;
  const int qr   = tid >> 4;            // Q-tile row/col for this thread
  const int qc   = (tid & 15) << 3;

  // K tile: 128x128 bf16 = 2048 uint4, eight per thread
#pragma unroll
  for (int i = tid; i < 2048; i += 256) {
    int r = i >> 4, c = (i & 15) << 3;
    *(uint4*)&Ks[r][c] = *(const uint4*)&kb[(size_t)(k0 + r) * 128 + c];
  }
  for (int i = tid; i < 16 * 32; i += 256)
    Ws[i >> 5][i & 31] = wts[(size_t)(q0 + (i >> 5)) * NHEAD + (i & 31)];

  uint4 qReg = *(const uint4*)&qb[((size_t)(q0 + qr)) * (NHEAD * DHEAD) + qc];

  int buf = 0;
  float out[8] = {};
  for (int h = 0; h < NHEAD; ++h) {
    *(uint4*)&Qs[buf][qr][qc] = qReg;
    __syncthreads();  // first iteration also covers Ks/Ws fill

    if (h + 1 < NHEAD)   // prefetch next head's Q tile, overlapped with WMMAs
      qReg = *(const uint4*)&qb[((size_t)(q0 + qr)) * (NHEAD * DHEAD) +
                                (h + 1) * DHEAD + qc];

    v8f acc = {};
#pragma unroll
    for (int kk = 0; kk < 128; kk += 32) {
      v16bf af, bfr;
#pragma unroll
      for (int e = 0; e < 16; ++e) {
        int ka = (e < 8 ? e : e + 8) + lh * 8;
        af[e]  = Qs[buf][l16][kk + ka];
        bfr[e] = Ks[wave * 16 + l16][kk + ka];
      }
      acc = __builtin_amdgcn_wmma_f32_16x16x32_bf16(false, af, false, bfr,
                                                    (short)0, acc, false, false);
    }
#pragma unroll
    for (int e = 0; e < 8; ++e) {
      float sc = acc[e];
      sc = sc > 0.0f ? sc : 0.0f;           // relu
      out[e] += sc * Ws[e + lh * 8][h];     // head-weighted reduce
    }
    buf ^= 1;
  }
#pragma unroll
  for (int e = 0; e < 8; ++e)
    scores[(size_t)(q0 + e + lh * 8) * S_LEN + k0 + wave * 16 + l16] = out[e];
}

// ---------------------------------------------------------------------------
// Top-k: per query row, full bitonic sort of 2048 (value desc, idx asc on
// ties, matching jax.lax.top_k), write first 1024 indices.
// ---------------------------------------------------------------------------
__global__ __launch_bounds__(256) void topk_kernel(
    const float* __restrict__ scores, int* __restrict__ out)
{
  __shared__ float v[2048];
  __shared__ int   ix[2048];
  const int q = blockIdx.x, tid = threadIdx.x;

#pragma unroll
  for (int i = tid; i < 512; i += 256) {
    float4 v4 = *(const float4*)&scores[(size_t)q * 2048 + i * 4];
    v[i * 4 + 0] = v4.x; v[i * 4 + 1] = v4.y;
    v[i * 4 + 2] = v4.z; v[i * 4 + 3] = v4.w;
    ix[i * 4 + 0] = i * 4 + 0; ix[i * 4 + 1] = i * 4 + 1;
    ix[i * 4 + 2] = i * 4 + 2; ix[i * 4 + 3] = i * 4 + 3;
  }
  __syncthreads();

  for (int k = 2; k <= 2048; k <<= 1) {
    for (int j = k >> 1; j > 0; j >>= 1) {
      for (int i = tid; i < 2048; i += 256) {
        int p = i ^ j;
        if (p > i) {
          bool descend = ((i & k) == 0);
          float va = v[i], vb = v[p];
          int   ia = ix[i], ib = ix[p];
          bool aFirst = (va > vb) || (va == vb && ia < ib);
          bool doSwap = descend ? !aFirst : aFirst;
          if (doSwap) { v[i] = vb; v[p] = va; ix[i] = ib; ix[p] = ia; }
        }
      }
      __syncthreads();
    }
  }
  for (int i = tid; i < 1024; i += 256)
    out[(size_t)q * 1024 + i] = ix[i];
}

// ---------------------------------------------------------------------------
extern "C" void kernel_launch(void* const* d_in, const int* in_sizes, int n_in,
                              void* d_out, int out_size, void* d_ws, size_t ws_size,
                              hipStream_t stream) {
  const float* x         = (const float*)d_in[0];  // (1, 2048, 4096)
  const float* q_resid   = (const float*)d_in[1];  // (1, 2048, 1536)
  const float* freqs     = (const float*)d_in[2];  // (2048, 32)
  const float* wq_b      = (const float*)d_in[3];  // (4096, 1536)
  const float* wk        = (const float*)d_in[4];  // (128, 4096)
  const float* w_weights = (const float*)d_in[5];  // (32, 4096)
  const float* kn_gamma  = (const float*)d_in[6];  // (128,)
  const float* kn_beta   = (const float*)d_in[7];  // (128,)
  int* out = (int*)d_out;                          // (1, 2048, 1024) int32

  // Workspace layout (~68 MB)
  char* ws = (char*)d_ws;
  float*  q_f32  = (float*)(ws);                          // 32 MB
  float*  k_f32  = (float*)(ws + (33ull << 20));          //  1 MB
  float*  wgt    = (float*)(ws + (34ull << 20));          // 256 KB
  float*  scores = (float*)(ws + (35ull << 20));          // 16 MB
  __bf16* qb     = (__bf16*)(ws + (51ull << 20));         // 16 MB
  __bf16* kb     = (__bf16*)(ws + (67ull << 20));         // 512 KB

  const dim3 blk(256);

  // 1) Projections (bf16 WMMA GEMMs)
  gemm_bf16_nt<<<dim3(S_LEN / 16, (NHEAD * DHEAD) / 128), blk, 0, stream>>>(
      q_resid, wq_b, q_f32, S_LEN, NHEAD * DHEAD, QLRANK, 1.0f);
  gemm_bf16_nt<<<dim3(S_LEN / 16, 1), blk, 0, stream>>>(
      x, wk, k_f32, S_LEN, DHEAD, HIDDEN, 1.0f);
  gemm_bf16_nt<<<dim3(S_LEN / 16, 1), blk, 0, stream>>>(
      x, w_weights, wgt, S_LEN, NHEAD, HIDDEN, 1.0f / 64.0f); // H^-.5 * D^-.5

  // 2) Epilogues: LN/RoPE/FWHT -> bf16 tensors
  k_epilogue<<<S_LEN, 128, 0, stream>>>(k_f32, freqs, kn_gamma, kn_beta, kb);
  q_epilogue<<<dim3(S_LEN, NHEAD), 128, 0, stream>>>(q_f32, freqs, qb);

  // 3) Score einsum + relu + head-weighted reduce (bf16 WMMA)
  scores_kernel<<<dim3(S_LEN / 16, S_LEN / 128), blk, 0, stream>>>(
      qb, kb, wgt, scores);

  // 4) Top-1024 indices per query
  topk_kernel<<<S_LEN, 256, 0, stream>>>(scores, out);
}
